// SimpleMemoryBank_850403525346
// MI455X (gfx1250) — matrix operand order repaired and checked by reading, not compiled, not run on hardware
//
#include <hip/hip_runtime.h>
#include <math.h>

typedef __attribute__((ext_vector_type(16))) __bf16 v16bf;
typedef __attribute__((ext_vector_type(8)))  float  v8f;
typedef __attribute__((ext_vector_type(4)))  float  f4;
typedef __attribute__((ext_vector_type(4)))  int    v4i;

#define TILE_M   64     // query rows per workgroup
#define TILE_S   128    // S (memory slots) == columns of score tile
#define DKC      128    // D-chunk staged in LDS per outer iteration
#define KSTR     136    // ushort stride for bf16 LDS tiles (pad: 272B = 17*16, conflict-free)
#define SCSTR    129    // float stride for score tile (conflict-free column scans)
#define TOPK     8

union FragU { uint4 u[2]; v16bf f; };

__device__ __forceinline__ unsigned short bf16_rne(float x) {
    unsigned int u = __float_as_uint(x);
    u += 0x7FFFu + ((u >> 16) & 1u);
    return (unsigned short)(u >> 16);
}

// 8 floats -> packed bf16 hi / lo planes
__device__ __forceinline__ void split8(const f4 a, const f4 b, uint4& ph, uint4& pl) {
    float xs[8] = {a[0], a[1], a[2], a[3], b[0], b[1], b[2], b[3]};
    unsigned short h[8], l[8];
    #pragma unroll
    for (int e = 0; e < 8; ++e) {
        h[e] = bf16_rne(xs[e]);
        l[e] = bf16_rne(xs[e] - __uint_as_float((unsigned)h[e] << 16));
    }
    ph = make_uint4((unsigned)h[0] | ((unsigned)h[1] << 16), (unsigned)h[2] | ((unsigned)h[3] << 16),
                    (unsigned)h[4] | ((unsigned)h[5] << 16), (unsigned)h[6] | ((unsigned)h[7] << 16));
    pl = make_uint4((unsigned)l[0] | ((unsigned)l[1] << 16), (unsigned)l[2] | ((unsigned)l[3] << 16),
                    (unsigned)l[4] | ((unsigned)l[5] << 16), (unsigned)l[6] | ((unsigned)l[7] << 16));
}

// Async global->LDS 16B copy (CDNA5 GLOBAL_LOAD_ASYNC_TO_LDS_B128, ASYNCcnt)
__device__ __forceinline__ void async_copy16(void* lds, const void* gptr) {
#if defined(__gfx1250__) && __has_builtin(__builtin_amdgcn_global_load_async_to_lds_b128)
    __builtin_amdgcn_global_load_async_to_lds_b128(
        (__attribute__((address_space(1))) v4i*)gptr,
        (__attribute__((address_space(3))) v4i*)lds, 0, 0);
#else
    *(uint4*)lds = *(const uint4*)gptr;
#endif
}

__device__ __forceinline__ void wait_async0() {
#if defined(__gfx1250__) && __has_builtin(__builtin_amdgcn_s_wait_asynccnt)
    __builtin_amdgcn_s_wait_asynccnt(0);
#elif defined(__AMDGCN__)
    asm volatile("s_wait_asynccnt 0" ::: "memory");
#endif
}

// ---------------- Kernel A: K f32 -> bf16 hi/lo planes in workspace ----------------
__global__ void smb_convert_kernel(const float* __restrict__ Kw,
                                   unsigned short* __restrict__ khi,
                                   unsigned short* __restrict__ klo, int n) {
    int i = blockIdx.x * blockDim.x + threadIdx.x;
    if (i < n) {
        float x = Kw[i];
        unsigned short h = bf16_rne(x);
        float hf = __uint_as_float((unsigned int)h << 16);
        khi[i] = h;
        klo[i] = bf16_rne(x - hf);
    }
}

// ---------------- Kernel B: scores (WMMA) + top-8 + softmax + V gather -------------
__launch_bounds__(256, 2)
__global__ void smb_main_kernel(const float* __restrict__ q,
                                const unsigned short* __restrict__ khi,
                                const unsigned short* __restrict__ klo,
                                const float* __restrict__ V,
                                const float* __restrict__ sal,
                                float* __restrict__ out_read,  // [BT, D]
                                float* __restrict__ out_w,     // [BT, TOPK]
                                int BT, int D, int S) {
    __shared__ __align__(16) unsigned short sQh[TILE_M * KSTR];
    __shared__ __align__(16) unsigned short sQl[TILE_M * KSTR];
    __shared__ __align__(16) unsigned short sKh[TILE_S * KSTR];
    __shared__ __align__(16) unsigned short sKl[TILE_S * KSTR];
    __shared__ __align__(16) float sScore[TILE_M * SCSTR];
    __shared__ float sW[TILE_M][TOPK];
    __shared__ int   sIdx[TILE_M][TOPK];

    const int tid  = threadIdx.x;
    const int wave = tid >> 5;          // 0..7  -> S column block
    const int lane = tid & 31;
    const int hl   = lane & 15;         // position within half-wave
    const int row0 = blockIdx.x * TILE_M;

    v8f acc[4];
    #pragma unroll
    for (int m = 0; m < 4; ++m)
        #pragma unroll
        for (int i = 0; i < 8; ++i) acc[m][i] = 0.0f;

    // ---------- Phase 1: score GEMM over D in chunks of DKC ----------
    for (int d0 = 0; d0 < D; d0 += DKC) {
        // K tiles (pre-converted bf16 planes): pure copy -> async to-LDS, no VGPR round trip
        #pragma unroll
        for (int j = 0; j < 8; ++j) {
            int idx = j * 256 + tid;            // 2048 x 16B per plane
            int r   = idx >> 4;                 // 16 uint4 per row
            int c   = idx & 15;
            async_copy16(&sKh[r * KSTR + c * 8], (const uint4*)(khi + (size_t)r * D + d0) + c);
            async_copy16(&sKl[r * KSTR + c * 8], (const uint4*)(klo + (size_t)r * D + d0) + c);
        }

        // q tile: load all (non-temporal, read-once stream), then convert+store
        f4 qa[4], qb[4];
        #pragma unroll
        for (int j = 0; j < 4; ++j) {
            int idx = j * 256 + tid;            // 1024 chunks of 8 floats
            int r   = idx >> 4;                 // 16 chunks per row
            int c8  = idx & 15;
            const f4* gp = (const f4*)(q + (size_t)(row0 + r) * D + d0 + c8 * 8);
            qa[j] = __builtin_nontemporal_load(gp);
            qb[j] = __builtin_nontemporal_load(gp + 1);
        }
        #pragma unroll
        for (int j = 0; j < 4; ++j) {
            int idx = j * 256 + tid;
            int r   = idx >> 4;
            int c8  = idx & 15;
            uint4 ph, pl;
            split8(qa[j], qb[j], ph, pl);
            *(uint4*)&sQh[r * KSTR + c8 * 8] = ph;
            *(uint4*)&sQl[r * KSTR + c8 * 8] = pl;
        }
        wait_async0();
        __syncthreads();

        #pragma unroll
        for (int kc = 0; kc < DKC; kc += 32) {
            // B fragments: lane hl -> column n; low half K=kc..kc+15, high half +16
            int nrow = wave * 16 + hl;
            int kb   = kc + ((lane < 16) ? 0 : 16);
            const unsigned short* pbh = &sKh[nrow * KSTR + kb];
            const unsigned short* pbl = &sKl[nrow * KSTR + kb];
            FragU bh, bl;
            bh.u[0] = *(const uint4*)pbh;  bh.u[1] = *(const uint4*)(pbh + 8);
            bl.u[0] = *(const uint4*)pbl;  bl.u[1] = *(const uint4*)(pbl + 8);

            #pragma unroll
            for (int m = 0; m < 4; ++m) {
                // A layout: low half K=k..k+15, high half +16
                int arow = m * 16 + hl;
                int ka   = kc + ((lane < 16) ? 0 : 16);
                const unsigned short* pah = &sQh[arow * KSTR + ka];
                const unsigned short* pal = &sQl[arow * KSTR + ka];
                FragU ah, al;
                ah.u[0] = *(const uint4*)pah;  ah.u[1] = *(const uint4*)(pah + 8);
                al.u[0] = *(const uint4*)pal;  al.u[1] = *(const uint4*)(pal + 8);
                // bf16 split-precision: hi*hi + hi*lo + lo*hi into f32 accumulator
                acc[m] = __builtin_amdgcn_wmma_f32_16x16x32_bf16(false, ah.f, false, bh.f,
                                                                 (short)0, acc[m], false, false);
                acc[m] = __builtin_amdgcn_wmma_f32_16x16x32_bf16(false, ah.f, false, bl.f,
                                                                 (short)0, acc[m], false, false);
                acc[m] = __builtin_amdgcn_wmma_f32_16x16x32_bf16(false, al.f, false, bh.f,
                                                                 (short)0, acc[m], false, false);
            }
        }
        __syncthreads();
    }

    // ---------- Phase 2: scale + salience, scatter scores to LDS ----------
    const float scale = rsqrtf((float)D);
    const float sv    = sal[wave * 16 + hl];
    #pragma unroll
    for (int m = 0; m < 4; ++m) {
        int rbase = m * 16 + ((lane < 16) ? 0 : 8);
        #pragma unroll
        for (int i = 0; i < 8; ++i) {
            sScore[(rbase + i) * SCSTR + wave * 16 + hl] = acc[m][i] * scale + sv;
        }
    }
    __syncthreads();

    // ---------- Phase 3: per-row top-8 (ties -> lowest index), softmax ----------
    if (tid < TILE_M) {
        float* rowp = &sScore[tid * SCSTR];
        float vals[TOPK]; int idxs[TOPK];
        #pragma unroll
        for (int kk = 0; kk < TOPK; ++kk) {
            float best = -__builtin_inff(); int bi = 0;
            for (int s = 0; s < S; ++s) {
                float x = rowp[s];
                if (x > best) { best = x; bi = s; }
            }
            vals[kk] = best; idxs[kk] = bi;
            rowp[bi] = -__builtin_inff();
        }
        float mx = vals[0], sum = 0.0f, w[TOPK];
        #pragma unroll
        for (int kk = 0; kk < TOPK; ++kk) { w[kk] = expf(vals[kk] - mx); sum += w[kk]; }
        float inv = 1.0f / sum;
        size_t rg = (size_t)(row0 + tid);
        #pragma unroll
        for (int kk = 0; kk < TOPK; ++kk) {
            float ww = w[kk] * inv;
            sW[tid][kk] = ww; sIdx[tid][kk] = idxs[kk];
            __builtin_nontemporal_store(ww, &out_w[rg * TOPK + kk]);
        }
    }
    __syncthreads();

    // ---------- Phase 4: weighted gather of V rows (L2-resident), coalesced ----------
    const int c4n = D >> 2;   // float4 per row
    for (int r = 0; r < TILE_M; ++r) {
        float ww[TOPK]; int sid[TOPK];
        #pragma unroll
        for (int kk = 0; kk < TOPK; ++kk) { ww[kk] = sW[r][kk]; sid[kk] = sIdx[r][kk]; }
        for (int c4 = tid; c4 < c4n; c4 += 256) {
            f4 a = {0.f, 0.f, 0.f, 0.f};
            #pragma unroll
            for (int kk = 0; kk < TOPK; ++kk) {
                f4 vv = ((const f4*)(V + (size_t)sid[kk] * D))[c4];
                a += vv * ww[kk];
            }
            __builtin_nontemporal_store(a, (f4*)(out_read + (size_t)(row0 + r) * D) + c4);
        }
    }
}

extern "C" void kernel_launch(void* const* d_in, const int* in_sizes, int n_in,
                              void* d_out, int out_size, void* d_ws, size_t ws_size,
                              hipStream_t stream) {
    const float* q   = (const float*)d_in[0];
    const float* Kw  = (const float*)d_in[1];
    const float* V   = (const float*)d_in[2];
    const float* sal = (const float*)d_in[3];

    const int S  = in_sizes[3];             // 128
    const int D  = in_sizes[1] / S;         // 1024
    const int BT = in_sizes[0] / D;         // 32768

    unsigned short* khi = (unsigned short*)d_ws;
    unsigned short* klo = khi + (size_t)S * D;

    float* out_read = (float*)d_out;                     // [BT, D]
    float* out_w    = out_read + (size_t)BT * D;         // [BT, TOPK]

    const int n = S * D;
    smb_convert_kernel<<<(n + 255) / 256, 256, 0, stream>>>(Kw, khi, klo, n);
    smb_main_kernel<<<BT / TILE_M, 256, 0, stream>>>(q, khi, klo, V, sal,
                                                     out_read, out_w, BT, D, S);
}
